// SpeakerModelHistAttCopy_84559316124103
// MI455X (gfx1250) — matrix operand order repaired and checked by reading, not compiled, not run on hardware
//
#include <hip/hip_runtime.h>
#include <hip/hip_bf16.h>
#include <math.h>

// ---------------------------------------------------------------------------
// Model dims (fixed by the reference)
// ---------------------------------------------------------------------------
#define BB      64
#define T_UTT   41
#define TDEC    40
#define L_PREV  64
#define N_UNK   20
#define VV      8191
#define V_EXT   8211
#define EE      512
#define HH      512
#define AA      512
#define IMG     2048

typedef __attribute__((ext_vector_type(16))) _Float16 v16h;
typedef __attribute__((ext_vector_type(8)))  float    v8f;

__device__ __forceinline__ float sigmoidf_(float x) { return 1.0f / (1.0f + __expf(-x)); }

// ---------------------------------------------------------------------------
// Generic dual-segment WMMA GEMM:  Y[M,N] = act( X1@W1^T + X2@W2^T + bias )
//   X row-major with row stride ldx; W row-major (N,K) with row stride ldw.
//   f32 -> f16 conversion on load, f32 accumulate via v_wmma_f32_16x16x32_f16.
//   Block = 128 threads (4 waves); block tile 32(M) x 64(N).
//   Each wave owns a 16x32 C tile: one A fragment is reused against two B
//   fragments per K-step (2 WMMAs per fragment-load round -> halved A traffic).
//   K1,K2 must be multiples of 32 (true for all uses here).
// ---------------------------------------------------------------------------
__global__ __launch_bounds__(128)
void k_gemm(const float* __restrict__ X1, int ldx1,
            const float* __restrict__ W1, int ldw1, int K1,
            const float* __restrict__ X2, int ldx2,
            const float* __restrict__ W2, int ldw2, int K2,
            const float* __restrict__ bias,
            float* __restrict__ Y, int ldy,
            int M, int N, int act)
{
    const int lane = threadIdx.x & 31;
    const int wv   = threadIdx.x >> 5;            // 0..3
    const int m0   = blockIdx.y * 32 + (wv >> 1) * 16;
    const int n0   = blockIdx.x * 64 + (wv & 1) * 32;   // wave covers n0..n0+31
    const int half = lane >> 4;                   // 0/1
    const int ml   = lane & 15;

    v8f acc0 = {};
    v8f acc1 = {};

    const float* Xp = X1; const float* Wp = W1;
    int K = K1, ldx = ldx1, ldw = ldw1;
    for (int seg = 0; seg < 2; ++seg) {
        if (seg == 1) { Xp = X2; Wp = W2; K = K2; ldx = ldx2; ldw = ldw2; }
        if (Xp == nullptr || K <= 0) continue;

        int arow  = m0 + ml;      if (arow  >= M) arow  = M - 1;
        int brow0 = n0 + ml;      if (brow0 >= N) brow0 = N - 1;
        int brow1 = n0 + 16 + ml; if (brow1 >= N) brow1 = N - 1;
        const float* aptr  = Xp + (size_t)arow  * ldx;
        const float* bptr0 = Wp + (size_t)brow0 * ldw;
        const float* bptr1 = Wp + (size_t)brow1 * ldw;

        for (int kb = 0; kb < K; kb += 32) {
            __builtin_prefetch(aptr  + kb + 128, 0, 1);
            __builtin_prefetch(bptr0 + kb + 128, 0, 1);
            __builtin_prefetch(bptr1 + kb + 128, 0, 1);
            v16h a, b0, b1;
            // A fragment: 16x32 f16, lane m = ml, k = e + 16*half
            #pragma unroll
            for (int e = 0; e < 16; e += 2) {
                int j  = e >> 1;
                int k0 = 16 * half + 2 * j;
                a[e]     = (_Float16)aptr[kb + k0];
                a[e + 1] = (_Float16)aptr[kb + k0 + 1];
            }
            // B fragments: 32x16 f16 (B = W^T), k = e + 16*half, n = ml
            #pragma unroll
            for (int e = 0; e < 16; ++e) {
                b0[e] = (_Float16)bptr0[kb + 16 * half + e];
                b1[e] = (_Float16)bptr1[kb + 16 * half + e];
            }
            acc0 = __builtin_amdgcn_wmma_f32_16x16x32_f16(
                false, a, false, b0, (short)0, acc0, false, false);
            acc1 = __builtin_amdgcn_wmma_f32_16x16x32_f16(
                false, a, false, b1, (short)0, acc1, false, false);
        }
    }

    #pragma unroll
    for (int r = 0; r < 8; ++r) {
        int m = m0 + r + 8 * half;
        if (m >= M) continue;
        int n = n0 + ml;
        if (n < N) {
            float v = acc0[r] + (bias ? bias[n] : 0.0f);
            if (act == 1)      v = fmaxf(v, 0.0f);
            else if (act == 2) v = tanhf(v);
            Y[(size_t)m * ldy + n] = v;
        }
        int n2 = n + 16;
        if (n2 < N) {
            float v = acc1[r] + (bias ? bias[n2] : 0.0f);
            if (act == 1)      v = fmaxf(v, 0.0f);
            else if (act == 2) v = tanhf(v);
            Y[(size_t)m * ldy + n2] = v;
        }
    }
}

// ---------------------------------------------------------------------------
// Embedding gathers (forward + length-reversed) for the encoder
// ---------------------------------------------------------------------------
__global__ void k_setup_embeds(const float* __restrict__ emb,
                               const int* __restrict__ prev_utt,
                               const int* __restrict__ lens,
                               float* __restrict__ embeds,
                               float* __restrict__ embeds_rev)
{
    int b = blockIdx.x >> 6, t = blockIdx.x & 63, e = threadIdx.x;  // 512 thr
    int tok = prev_utt[b * L_PREV + t];
    embeds[((size_t)b * L_PREV + t) * EE + e] = emb[(size_t)tok * EE + e];
    int len = lens[b];
    int ri = len - 1 - t; ri = ri < 0 ? 0 : (ri > L_PREV - 1 ? L_PREV - 1 : ri);
    int tok2 = prev_utt[b * L_PREV + ri];
    embeds_rev[((size_t)b * L_PREV + t) * EE + e] = emb[(size_t)tok2 * EE + e];
}

// tgt_emb gather + dec_inp_gen = tanh(lin2pgen_decinp(tgt_emb))
__global__ void k_tgt_emb(const float* __restrict__ emb,
                          const int* __restrict__ utt,
                          const float* __restrict__ wdig,
                          const float* __restrict__ bdig,
                          float* __restrict__ tgt_emb,
                          float* __restrict__ dig)
{
    int bt = blockIdx.x, tid = threadIdx.x;  // 256 thr
    int tok = utt[bt];
    __shared__ float red[256];
    float acc = 0.f;
    for (int e = tid; e < EE; e += 256) {
        float v = emb[(size_t)tok * EE + e];
        tgt_emb[(size_t)bt * EE + e] = v;
        acc += wdig[e] * v;
    }
    red[tid] = acc; __syncthreads();
    for (int s = 128; s > 0; s >>= 1) { if (tid < s) red[tid] += red[tid + s]; __syncthreads(); }
    if (tid == 0) dig[bt] = tanhf(red[0] + bdig[0]);
}

// ---------------------------------------------------------------------------
// LSTM pointwise: gates (B,4H) -> update h,c (masked), optional seq output
// ---------------------------------------------------------------------------
__global__ void k_lstm_point(const float* __restrict__ gates,
                             float* __restrict__ h, float* __restrict__ c,
                             float* __restrict__ out_seq, int t,
                             const int* __restrict__ lens)
{
    int b = blockIdx.x, j = threadIdx.x;  // 512 thr
    const float* g = gates + (size_t)b * (4 * HH);
    float gi = sigmoidf_(g[j]);
    float gf = sigmoidf_(g[HH + j]);
    float gg = tanhf(g[2 * HH + j]);
    float go = sigmoidf_(g[3 * HH + j]);
    float c2 = gf * c[(size_t)b * HH + j] + gi * gg;
    float h2 = go * tanhf(c2);
    bool m = lens ? (t < lens[b]) : true;
    if (m) { h[(size_t)b * HH + j] = h2; c[(size_t)b * HH + j] = c2; }
    if (out_seq) out_seq[((size_t)b * L_PREV + t) * HH + j] = m ? h2 : 0.0f;
}

// outputs[b,t,:H]=out_f ; outputs[b,t,H:]= valid ? out_r[b,clip(len-1-t)] : 0
__global__ void k_build_outputs(const float* __restrict__ out_f,
                                const float* __restrict__ out_r,
                                const int* __restrict__ lens,
                                float* __restrict__ outputs)
{
    int b = blockIdx.x >> 6, t = blockIdx.x & 63, j = threadIdx.x;  // 512 thr
    int len = lens[b];
    float* o = outputs + ((size_t)b * L_PREV + t) * (2 * HH);
    o[j] = out_f[((size_t)b * L_PREV + t) * HH + j];
    int ri = len - 1 - t; ri = ri < 0 ? 0 : (ri > L_PREV - 1 ? L_PREV - 1 : ri);
    float v = out_r[((size_t)b * L_PREV + ri) * HH + j];
    o[HH + j] = (t < len) ? v : 0.0f;
}

__global__ void k_copy(float* __restrict__ dst, const float* __restrict__ src, int n)
{
    int i = blockIdx.x * blockDim.x + threadIdx.x;
    if (i < n) dst[i] = src[i];
}
__global__ void k_zero(float* __restrict__ p, int n)
{
    int i = blockIdx.x * blockDim.x + threadIdx.x;
    if (i < n) p[i] = 0.0f;
}

// ---------------------------------------------------------------------------
// Attention + coverage + p_gen (one block per batch row, 256 threads)
// ---------------------------------------------------------------------------
__global__ void k_attention(const float* __restrict__ hist_att,
                            const float* __restrict__ h1att,
                            const float* __restrict__ h1,
                            const float* __restrict__ covw,   // (A,1)
                            const float* __restrict__ covb,   // (A,)
                            const float* __restrict__ attw_v, // (1,A)
                            const float* __restrict__ att_b,  // (1,)
                            const float* __restrict__ encw, const float* __restrict__ encb,
                            const float* __restrict__ decw, const float* __restrict__ decb,
                            const float* __restrict__ dig, int t,
                            const int* __restrict__ lens,
                            float* __restrict__ cov, float* __restrict__ covloss,
                            float* __restrict__ ctx, float* __restrict__ pgen,
                            float* __restrict__ las)
{
    int b = blockIdx.x, tid = threadIdx.x;
    __shared__ float sc[L_PREV], w[L_PREV], red[256];
    __shared__ float s_max, s_sum, s_lse2;
    int len = lens[b];

    for (int l = 0; l < L_PREV; ++l) {
        float cv = cov[b * L_PREV + l];
        float p = 0.f;
        for (int a = tid; a < AA; a += 256) {
            float v = hist_att[((size_t)b * L_PREV + l) * AA + a]
                    + h1att[(size_t)b * AA + a] + cv * covw[a] + covb[a];
            p += attw_v[a] * tanhf(v);
        }
        red[tid] = p; __syncthreads();
        for (int s = 128; s > 0; s >>= 1) { if (tid < s) red[tid] += red[tid + s]; __syncthreads(); }
        if (tid == 0) sc[l] = (l < len) ? (red[0] + att_b[0]) : -1e30f;
        __syncthreads();
    }
    if (tid == 0) {
        float mx = -1e30f; for (int l = 0; l < L_PREV; ++l) mx = fmaxf(mx, sc[l]);
        float sm = 0.f;    for (int l = 0; l < L_PREV; ++l) sm += __expf(sc[l] - mx);
        s_max = mx; s_sum = sm;
    }
    __syncthreads();
    if (tid < L_PREV) w[tid] = __expf(sc[tid] - s_max) / s_sum;
    __syncthreads();
    // log_softmax over the (probability-valued) att weights
    if (tid == 0) {
        float mx = -1e30f; for (int l = 0; l < L_PREV; ++l) mx = fmaxf(mx, w[l]);
        float sm = 0.f;    for (int l = 0; l < L_PREV; ++l) sm += __expf(w[l] - mx);
        s_lse2 = mx + __logf(sm);
    }
    __syncthreads();
    if (tid < L_PREV) las[b * L_PREV + tid] = w[tid] - s_lse2;

    // ctx + p_gen dot partials
    float p1 = 0.f, p2 = 0.f;
    for (int a = tid; a < AA; a += 256) {
        float s = 0.f;
        for (int l = 0; l < L_PREV; ++l)
            s += hist_att[((size_t)b * L_PREV + l) * AA + a] * w[l];
        ctx[(size_t)b * AA + a] = s;
        p1 += encw[a] * s;
        p2 += decw[a] * h1[(size_t)b * HH + a];
    }
    red[tid] = p1; __syncthreads();
    for (int s = 128; s > 0; s >>= 1) { if (tid < s) red[tid] += red[tid + s]; __syncthreads(); }
    float d1 = red[0] + encb[0]; __syncthreads();
    red[tid] = p2; __syncthreads();
    for (int s = 128; s > 0; s >>= 1) { if (tid < s) red[tid] += red[tid + s]; __syncthreads(); }
    float d2 = red[0] + decb[0]; __syncthreads();
    if (tid == 0) {
        pgen[b] = sigmoidf_(tanhf(d1) + tanhf(d2) + dig[b * T_UTT + t]);
        float cl = 0.f;
        for (int l = 0; l < L_PREV; ++l) cl += fminf(w[l], cov[b * L_PREV + l]);
        covloss[b] += cl;
    }
    __syncthreads();
    if (tid < L_PREV) cov[b * L_PREV + tid] += w[tid];
}

// ---------------------------------------------------------------------------
// log-softmax over vocab, blend with p_gen, scatter pointer distribution
// ---------------------------------------------------------------------------
__global__ void k_final(const float* __restrict__ logits, // ld 8192
                        const float* __restrict__ pgen,
                        const float* __restrict__ las,
                        const int* __restrict__ mapped,
                        int t, float* __restrict__ preds)
{
    int b = blockIdx.x, tid = threadIdx.x;  // 256 thr
    __shared__ float red[256];
    __shared__ float s_mx, s_lse;
    const float* lg = logits + (size_t)b * 8192;
    float mx = -1e30f;
    for (int v = tid; v < VV; v += 256) mx = fmaxf(mx, lg[v]);
    red[tid] = mx; __syncthreads();
    for (int s = 128; s > 0; s >>= 1) { if (tid < s) red[tid] = fmaxf(red[tid], red[tid + s]); __syncthreads(); }
    if (tid == 0) s_mx = red[0]; __syncthreads();
    float sm = 0.f;
    for (int v = tid; v < VV; v += 256) sm += __expf(lg[v] - s_mx);
    red[tid] = sm; __syncthreads();
    for (int s = 128; s > 0; s >>= 1) { if (tid < s) red[tid] += red[tid + s]; __syncthreads(); }
    if (tid == 0) s_lse = s_mx + __logf(red[0]); __syncthreads();

    float pg = pgen[b];
    float* out = preds + ((size_t)b * TDEC + t) * V_EXT;
    for (int v = tid; v < V_EXT; v += 256)
        out[v] = (v < VV) ? pg * (lg[v] - s_lse) : 0.0f;
    __syncthreads();
    if (tid == 0) {
        float om = 1.0f - pg;
        for (int l = 0; l < L_PREV; ++l)
            out[mapped[b * L_PREV + l]] += om * las[b * L_PREV + l];
    }
}

// ---------------------------------------------------------------------------
// Host orchestration
// ---------------------------------------------------------------------------
static inline void gemm(hipStream_t s,
                        const float* X1, int ldx1, const float* W1, int ldw1, int K1,
                        const float* X2, int ldx2, const float* W2, int ldw2, int K2,
                        const float* bias, float* Y, int ldy, int M, int N, int act)
{
    dim3 g((N + 63) / 64, (M + 31) / 32), b(128);
    k_gemm<<<g, b, 0, s>>>(X1, ldx1, W1, ldw1, K1, X2, ldx2, W2, ldw2, K2,
                           bias, Y, ldy, M, N, act);
}

extern "C" void kernel_launch(void* const* d_in, const int* in_sizes, int n_in,
                              void* d_out, int out_size, void* d_ws, size_t ws_size,
                              hipStream_t stream)
{
    (void)in_sizes; (void)n_in; (void)out_size; (void)ws_size;
    const float* emb        = (const float*)d_in[0];
    const float* hid_w      = (const float*)d_in[1];  const float* hid_b    = (const float*)d_in[2];
    const float* dec_l_w    = (const float*)d_in[3];  const float* dec_l_b  = (const float*)d_in[4];
    const float* visc_w     = (const float*)d_in[5];  const float* visc_b   = (const float*)d_in[6];
    const float* sep_w      = (const float*)d_in[7];  const float* sep_b    = (const float*)d_in[8];
    const float* a_hid_w    = (const float*)d_in[9];  const float* a_hid_b  = (const float*)d_in[10];
    const float* a_hist_w   = (const float*)d_in[11]; const float* a_hist_b = (const float*)d_in[12];
    const float* h2m_w      = (const float*)d_in[13]; const float* h2m_b    = (const float*)d_in[14];
    const float* m2v_w      = (const float*)d_in[15]; const float* m2v_b    = (const float*)d_in[16];
    const float* pg_enc_w   = (const float*)d_in[17]; const float* pg_enc_b = (const float*)d_in[18];
    const float* pg_dinp_w  = (const float*)d_in[19]; const float* pg_dinp_b= (const float*)d_in[20];
    const float* pg_dst_w   = (const float*)d_in[21]; const float* pg_dst_b = (const float*)d_in[22];
    const float* cov_w      = (const float*)d_in[23]; const float* cov_b    = (const float*)d_in[24];
    const float* att_w      = (const float*)d_in[25]; const float* att_b    = (const float*)d_in[26];
    const float* enc_f_wi   = (const float*)d_in[27]; const float* enc_f_wh = (const float*)d_in[28];
    const float* enc_f_b    = (const float*)d_in[29];
    const float* enc_b_wi   = (const float*)d_in[30]; const float* enc_b_wh = (const float*)d_in[31];
    const float* enc_b_b    = (const float*)d_in[32];
    const float* dec_wi     = (const float*)d_in[33]; const float* dec_wh   = (const float*)d_in[34];
    const float* dec_b      = (const float*)d_in[35];
    const float* vis_ctx    = (const float*)d_in[36];
    const float* tgt_feats  = (const float*)d_in[37];
    const int*   utter      = (const int*)d_in[38];
    const int*   prev_utt   = (const int*)d_in[39];
    const int*   lens       = (const int*)d_in[40];
    const int*   mapped     = (const int*)d_in[43];

    float* preds = (float*)d_out;  // (B,TDEC,V_EXT) then covloss (B,)

    // ---- workspace carve (deterministic) ----
    float* p = (float*)d_ws;
    auto alloc = [&](size_t n) { float* r = p; p += n; return r; };
    float* vis_hid    = alloc((size_t)BB * HH);
    float* tgt_hid    = alloc((size_t)BB * HH);
    float* cvi        = alloc((size_t)BB * HH);
    float* embeds     = alloc((size_t)BB * L_PREV * EE);
    float* embeds_rev = alloc((size_t)BB * L_PREV * EE);
    float* out_f      = alloc((size_t)BB * L_PREV * HH);
    float* out_r      = alloc((size_t)BB * L_PREV * HH);
    float* outputs    = alloc((size_t)BB * L_PREV * 2 * HH);
    float* hist_att   = alloc((size_t)BB * L_PREV * AA);
    float* tgt_emb    = alloc((size_t)BB * T_UTT * EE);
    float* dig        = alloc((size_t)BB * T_UTT);
    float* h_f        = alloc((size_t)BB * HH);
    float* c_f        = alloc((size_t)BB * HH);
    float* h_b        = alloc((size_t)BB * HH);
    float* c_b        = alloc((size_t)BB * HH);
    float* h1         = alloc((size_t)BB * HH);
    float* c1         = alloc((size_t)BB * HH);
    float* gates      = alloc((size_t)BB * 4 * HH);
    float* h1att      = alloc((size_t)BB * AA);
    float* las        = alloc((size_t)BB * L_PREV);
    float* ctx        = alloc((size_t)BB * AA);
    float* pgen       = alloc((size_t)BB);
    float* cov        = alloc((size_t)BB * L_PREV);
    float* covloss    = alloc((size_t)BB);
    float* mid        = alloc((size_t)BB * HH);
    float* logits     = alloc((size_t)BB * 8192);

    // ---- stage 0: visual features ----
    gemm(stream, vis_ctx, IMG * 6, visc_w, IMG * 6, IMG * 6,
         nullptr, 0, nullptr, 0, 0, visc_b, vis_hid, HH, BB, HH, 1);
    gemm(stream, tgt_feats, IMG, sep_w, IMG, IMG,
         nullptr, 0, nullptr, 0, 0, sep_b, tgt_hid, HH, BB, HH, 1);
    gemm(stream, vis_hid, HH, hid_w, 2 * HH, HH,
         tgt_hid, HH, hid_w + HH, 2 * HH, HH, hid_b, cvi, HH, BB, HH, 1);

    // ---- embeddings ----
    k_setup_embeds<<<BB * L_PREV, 512, 0, stream>>>(emb, prev_utt, lens, embeds, embeds_rev);
    k_tgt_emb<<<BB * T_UTT, 256, 0, stream>>>(emb, utter, pg_dinp_w, pg_dinp_b, tgt_emb, dig);

    // ---- init LSTM states h0 = c0 = cvi ----
    k_copy<<<(BB * HH + 255) / 256, 256, 0, stream>>>(h_f, cvi, BB * HH);
    k_copy<<<(BB * HH + 255) / 256, 256, 0, stream>>>(c_f, cvi, BB * HH);
    k_copy<<<(BB * HH + 255) / 256, 256, 0, stream>>>(h_b, cvi, BB * HH);
    k_copy<<<(BB * HH + 255) / 256, 256, 0, stream>>>(c_b, cvi, BB * HH);

    // ---- bidirectional encoder: gates = x@Wi^T + h@Wh^T + b (one dual GEMM) ----
    for (int t = 0; t < L_PREV; ++t) {
        gemm(stream, embeds + (size_t)t * EE, L_PREV * EE, enc_f_wi, EE, EE,
             h_f, HH, enc_f_wh, HH, HH, enc_f_b, gates, 4 * HH, BB, 4 * HH, 0);
        k_lstm_point<<<BB, 512, 0, stream>>>(gates, h_f, c_f, out_f, t, lens);
        gemm(stream, embeds_rev + (size_t)t * EE, L_PREV * EE, enc_b_wi, EE, EE,
             h_b, HH, enc_b_wh, HH, HH, enc_b_b, gates, 4 * HH, BB, 4 * HH, 0);
        k_lstm_point<<<BB, 512, 0, stream>>>(gates, h_b, c_b, out_r, t, lens);
    }
    k_build_outputs<<<BB * L_PREV, 512, 0, stream>>>(out_f, out_r, lens, outputs);

    // decoder_hid = linear_dec([h_f, h_b]); h1 = c1 = decoder_hid
    gemm(stream, h_f, HH, dec_l_w, 2 * HH, HH,
         h_b, HH, dec_l_w + HH, 2 * HH, HH, dec_l_b, h1, HH, BB, HH, 0);
    k_copy<<<(BB * HH + 255) / 256, 256, 0, stream>>>(c1, h1, BB * HH);

    // history_att = lin2att_hist(outputs), batched over B*L rows
    gemm(stream, outputs, 2 * HH, a_hist_w, 2 * HH, 2 * HH,
         nullptr, 0, nullptr, 0, 0, a_hist_b, hist_att, AA, BB * L_PREV, AA, 0);

    // coverage state
    k_zero<<<(BB * L_PREV + 255) / 256, 256, 0, stream>>>(cov, BB * L_PREV);
    k_zero<<<1, 64, 0, stream>>>(covloss, BB);

    // ---- decoder ----
    for (int t = 0; t < TDEC; ++t) {
        gemm(stream, tgt_emb + (size_t)t * EE, T_UTT * EE, dec_wi, EE, EE,
             h1, HH, dec_wh, HH, HH, dec_b, gates, 4 * HH, BB, 4 * HH, 0);
        k_lstm_point<<<BB, 512, 0, stream>>>(gates, h1, c1, nullptr, t, nullptr);

        gemm(stream, h1, HH, a_hid_w, HH, HH,
             nullptr, 0, nullptr, 0, 0, a_hid_b, h1att, AA, BB, AA, 0);

        k_attention<<<BB, 256, 0, stream>>>(hist_att, h1att, h1, cov_w, cov_b,
                                            att_w, att_b, pg_enc_w, pg_enc_b,
                                            pg_dst_w, pg_dst_b, dig, t, lens,
                                            cov, covloss, ctx, pgen, las);

        gemm(stream, h1, HH, h2m_w, AA + HH, HH,
             ctx, AA, h2m_w + HH, AA + HH, AA, h2m_b, mid, HH, BB, HH, 0);

        gemm(stream, mid, HH, m2v_w, HH, HH,
             nullptr, 0, nullptr, 0, 0, m2v_b, logits, 8192, BB, VV, 0);

        k_final<<<BB, 256, 0, stream>>>(logits, pgen, las, mapped, t, preds);
    }

    // covloss to tail of d_out
    k_copy<<<1, 64, 0, stream>>>(preds + (size_t)BB * TDEC * V_EXT, covloss, BB);
}
